// Experts_4037269258955
// MI455X (gfx1250) — matrix operand order repaired and checked by reading, not compiled, not run on hardware
//
#include <hip/hip_runtime.h>
#include <hip/hip_bf16.h>

typedef __attribute__((ext_vector_type(16))) __bf16 v16bf;
typedef __attribute__((ext_vector_type(8)))  __bf16 v8bf;
typedef __attribute__((ext_vector_type(8)))  float  v8f;

#define TT   4096      // tokens
#define DIMD 256       // output dims
#define NEXP 16        // experts
#define KDIM 768       // reduction dim (512 u + 256 R)
#define KP   776       // padded LDS row stride (388 dwords % 64 banks == 4 -> conflict-free)
#define MT   32        // token rows per block (2 WMMA tiles)

__device__ __forceinline__ __bf16 f2bf(float f) {
    unsigned u = __builtin_bit_cast(unsigned, f);
    u += 0x7FFFu + ((u >> 16) & 1u);           // round-to-nearest-even
    unsigned short h = (unsigned short)(u >> 16);
    return __builtin_bit_cast(__bf16, h);
}

// ---- Kernel 1: R[256] = cat(h, us, ue)[1280] @ Wr[1280,256] + br ----
__global__ __launch_bounds__(256)
void r_gemv_kernel(const float* __restrict__ h, const float* __restrict__ us,
                   const float* __restrict__ ue, const float* __restrict__ Wr,
                   const float* __restrict__ br, float* __restrict__ R) {
    int d = threadIdx.x;                        // 0..255
    float acc = br[d];
    for (int i = 0; i < 256; ++i) acc += h[i]  * Wr[(i      ) * DIMD + d];
    for (int i = 0; i < 512; ++i) acc += us[i] * Wr[(256 + i) * DIMD + d];
    for (int i = 0; i < 512; ++i) acc += ue[i] * Wr[(768 + i) * DIMD + d];
    R[d] = acc;
}

// ---- Kernel 2: xb[t][k] = bf16( k<512 ? u[t][k] : R[k-512] ) ----
__global__ __launch_bounds__(256)
void pack_x_kernel(const float* __restrict__ u, const float* __restrict__ R,
                   __bf16* __restrict__ xb) {
    int idx = blockIdx.x * 256 + threadIdx.x;   // TT*KDIM threads
    int t = idx / KDIM, c = idx - t * KDIM;
    float v = (c < 512) ? u[(size_t)t * 512 + c] : R[c - 512];
    xb[idx] = f2bf(v);
}

// ---- Kernel 3: Wt*[n][k] = bf16( W*[k][n] )  (transpose to [N][K]) ----
__global__ __launch_bounds__(256)
void pack_w_kernel(const float* __restrict__ Wn, const float* __restrict__ Wz,
                   const float* __restrict__ We, __bf16* __restrict__ wtn,
                   __bf16* __restrict__ wtz, __bf16* __restrict__ wte) {
    int idx = blockIdx.x * 256 + threadIdx.x;   // KDIM*4096 threads
    int kk = idx >> 12;                         // / 4096
    int n  = idx & 4095;
    size_t dst = (size_t)n * KDIM + kk;
    wtn[dst] = f2bf(Wn[idx]);
    wtz[dst] = f2bf(Wz[idx]);
    wte[dst] = f2bf(We[idx]);
}

// ---- Kernel 4: fused triple-GEMM (bf16 WMMA) + noisy top-2 gating epilogue ----
// Block = 8 waves x 2 token tiles. A (32 x 768) staged once in LDS and shared by
// all 8 waves; each wave owns one output dim (16 expert columns) and runs
// 6 WMMAs per K-step against a register-double-buffered B triple.
__global__ __launch_bounds__(256)
void moe_wmma_kernel(const __bf16* __restrict__ xb,
                     const __bf16* __restrict__ wtn,
                     const __bf16* __restrict__ wtz,
                     const __bf16* __restrict__ wte,
                     const float* __restrict__ bn, const float* __restrict__ bz,
                     const float* __restrict__ be, const float* __restrict__ noise,
                     float* __restrict__ out) {
    __shared__ __align__(16) __bf16 As[MT * KP];        // ~48.5 KB

    const int tid  = threadIdx.x;
    const int lane = tid & 31;
    const int wave = tid >> 5;
    const int m0   = blockIdx.x * MT;           // first token row of block
    const int d    = blockIdx.y * 8 + wave;     // output dim, 0..255
    const int n0   = d * NEXP;                  // 16 expert columns

    // ---- cooperative stage of A tile (32 rows x 768) into padded LDS ----
    {
        const int nchunks = MT * (KDIM / 8);    // 3072 x v8bf chunks
        for (int c = tid; c < nchunks; c += 256) {
            int row = c / (KDIM / 8);
            int col = c - row * (KDIM / 8);
            *(v8bf*)(As + row * KP + col * 8) =
                *(const v8bf*)(xb + (size_t)(m0 + row) * KDIM + col * 8);
        }
    }
    __syncthreads();

    const int mn   = lane & 15;                 // A-row / B-column within tile
    const int hi   = lane >> 4;
    const int aoff = hi ? 16 : 0;               // A 16x32 bf16 contiguous k = 16h + s
    const int boff = hi ? 16 : 0;               // B 32x16 bf16 contiguous k = 16h + s

    const __bf16* a0p   = As + mn * KP;                       // token tile 0
    const __bf16* a1p   = As + (16 + mn) * KP;                // token tile 1
    const __bf16* bnrow = wtn + (size_t)(n0 + mn) * KDIM + boff;
    const __bf16* bzrow = wtz + (size_t)(n0 + mn) * KDIM + boff;
    const __bf16* berow = wte + (size_t)(n0 + mn) * KDIM + boff;

    v8f cn[2] = {{}, {}}, cz[2] = {{}, {}}, ce[2] = {{}, {}};

    // prime B double buffer
    v16bf bmn = *(const v16bf*)(bnrow);
    v16bf bmz = *(const v16bf*)(bzrow);
    v16bf bme = *(const v16bf*)(berow);

    #pragma unroll 2
    for (int k = 0; k < KDIM; k += 32) {
        const int kn = (k + 32 < KDIM) ? (k + 32) : 0;        // keep loads unconditional
        __builtin_prefetch(bnrow + k + 256, 0, 1);
        v16bf bmn_n = *(const v16bf*)(bnrow + kn);
        v16bf bmz_n = *(const v16bf*)(bzrow + kn);
        v16bf bme_n = *(const v16bf*)(berow + kn);

        v16bf a0, a1;
        ((v8bf*)&a0)[0] = *(const v8bf*)(a0p + k + aoff);
        ((v8bf*)&a0)[1] = *(const v8bf*)(a0p + k + 8 + aoff);
        ((v8bf*)&a1)[0] = *(const v8bf*)(a1p + k + aoff);
        ((v8bf*)&a1)[1] = *(const v8bf*)(a1p + k + 8 + aoff);

        cn[0] = __builtin_amdgcn_wmma_f32_16x16x32_bf16(false, a0, false, bmn, (short)0, cn[0], false, false);
        cn[1] = __builtin_amdgcn_wmma_f32_16x16x32_bf16(false, a1, false, bmn, (short)0, cn[1], false, false);
        cz[0] = __builtin_amdgcn_wmma_f32_16x16x32_bf16(false, a0, false, bmz, (short)0, cz[0], false, false);
        cz[1] = __builtin_amdgcn_wmma_f32_16x16x32_bf16(false, a1, false, bmz, (short)0, cz[1], false, false);
        ce[0] = __builtin_amdgcn_wmma_f32_16x16x32_bf16(false, a0, false, bme, (short)0, ce[0], false, false);
        ce[1] = __builtin_amdgcn_wmma_f32_16x16x32_bf16(false, a1, false, bme, (short)0, ce[1], false, false);

        bmn = bmn_n; bmz = bmz_n; bme = bme_n;
    }

    // ---- Epilogue. C layout: VGPR r, lane -> token (base + r + 8*hi), expert (lane&15).
    const int eidx = lane & 15;
    const float bnb = bn[n0 + eidx];
    const float bzb = bz[n0 + eidx];
    const float beb = be[n0 + eidx];

    #pragma unroll
    for (int tt = 0; tt < 2; ++tt) {
        #pragma unroll
        for (int r = 0; r < 8; ++r) {
            const int t = m0 + tt * 16 + r + 8 * hi;
            const float noi = noise[(size_t)t * (DIMD * NEXP) + n0 + eidx]; // coalesced per half-wave
            const float lg  = (cn[tt][r] + bnb) + (cz[tt][r] + bzb) * noi;  // logit
            const float ev  = ce[tt][r] + beb;                              // expert output

            // top-2 over 16 experts (16-lane butterfly, tracks max + 2nd max)
            float m1 = lg, m2 = -3.0e38f;
            #pragma unroll
            for (int off = 8; off; off >>= 1) {
                float o1 = __shfl_xor(m1, off, 16);
                float o2 = __shfl_xor(m2, off, 16);
                float nh = fmaxf(m1, o1);
                float nl = fminf(m1, o1);
                m2 = fmaxf(nl, fmaxf(m2, o2));
                m1 = nh;
            }
            // softmax over kept experts (keep: logit >= 2nd max), gated sum
            float p   = (lg >= m2) ? __expf(lg - m1) : 0.0f;
            float num = p * ev;
            float den = p;
            #pragma unroll
            for (int off = 8; off; off >>= 1) {
                num += __shfl_xor(num, off, 16);
                den += __shfl_xor(den, off, 16);
            }
            if (eidx == 0) out[(size_t)t * DIMD + d] = num / den;
        }
    }
}

extern "C" void kernel_launch(void* const* d_in, const int* in_sizes, int n_in,
                              void* d_out, int out_size, void* d_ws, size_t ws_size,
                              hipStream_t stream) {
    const float* h     = (const float*)d_in[0];
    const float* us    = (const float*)d_in[1];
    const float* ue    = (const float*)d_in[2];
    const float* u     = (const float*)d_in[3];
    const float* noise = (const float*)d_in[4];
    const float* Wr    = (const float*)d_in[5];
    const float* br    = (const float*)d_in[6];
    const float* Wn    = (const float*)d_in[7];
    const float* bn    = (const float*)d_in[8];
    const float* Wz    = (const float*)d_in[9];
    const float* bz    = (const float*)d_in[10];
    const float* We    = (const float*)d_in[11];
    const float* be    = (const float*)d_in[12];
    float* out = (float*)d_out;

    // workspace layout: R (256 f32) | xb bf16 [4096][768] | Wt{n,z,e} bf16 [4096][768]
    float*  R   = (float*)d_ws;
    __bf16* xb  = (__bf16*)(R + 256);
    __bf16* wtn = xb  + (size_t)TT * KDIM;
    __bf16* wtz = wtn + (size_t)KDIM * 4096;
    __bf16* wte = wtz + (size_t)KDIM * 4096;

    r_gemv_kernel<<<1, 256, 0, stream>>>(h, us, ue, Wr, br, R);
    pack_x_kernel<<<(TT * KDIM) / 256, 256, 0, stream>>>(u, R, xb);
    pack_w_kernel<<<(KDIM * 4096) / 256, 256, 0, stream>>>(Wn, Wz, We, wtn, wtz, wte);

    dim3 grid(TT / MT, DIMD / 8);               // 128 token-tile groups x 32 dim-groups
    moe_wmma_kernel<<<grid, 256, 0, stream>>>(xb, wtn, wtz, wte, bn, bz, be, noise, out);
}
